// Deductron_86208583565662
// MI455X (gfx1250) — compile-verified, matrix operand order checked by reading, not run on hardware
//
#include <hip/hip_runtime.h>
#include <hip/hip_bf16.h>
#include <math.h>

#define N_MEMORY   128
#define INPUT_LEN  512
#define OUTPUT_LEN 256
#define N_FRAMES   65536
#define H_ROWS     256

#define FPB        32                       // frames per block == scan chunk
#define N_CHUNKS   (N_FRAMES / FPB)         // 2048
#define GRP_CH     32                       // chunks per carry group
#define N_GROUPS   (N_CHUNKS / GRP_CH)      // 64

typedef __attribute__((ext_vector_type(16))) _Float16 v16h;
typedef __attribute__((ext_vector_type(8)))  float    v8f;

union FragU { v16h v; unsigned u[8]; uint4 q[2]; _Float16 h[16]; };

__device__ __forceinline__ float sigmoidf_(float x) {
    return 1.0f / (1.0f + __expf(-x));
}

// K offset (within a 32-wide K step) of the uint element-pair j for a given
// lane half, per the CDNA5 16-bit A/B matrix VGPR layout.
__device__ __forceinline__ int kmap2(int j, int half) {
    return ((j < 4) ? (2 * j) : (2 * j + 8)) + 8 * half;
}

// Inverse of kmap2: even K-within-step kk -> (half, j)
__device__ __forceinline__ void kinv(int kk, int& h, int& j) {
    h = (kk >> 3) & 1;
    j = (kk >> 4) * 4 + ((kk & 7) >> 1);
}

// ---------------------------------------------------------------------------
// f32 weights (rows x K, row-major) -> fragment-ordered f16.
// ---------------------------------------------------------------------------
__global__ void k_w_to_frag(const float* __restrict__ W, unsigned* __restrict__ Wf,
                            int K, int nks, int total) {
    int idx = blockIdx.x * blockDim.x + threadIdx.x;
    if (idx >= total) return;
    int j    = idx & 7;
    int lane = (idx >> 3) & 31;
    int ks   = (idx >> 8) % nks;
    int rt   = (idx >> 8) / nks;
    int half = lane >> 4;
    int row  = rt * 16 + (lane & 15);
    int k    = ks * 32 + kmap2(j, half);
    union { _Float16 h[2]; unsigned u; } p;
    p.h[0] = (_Float16)W[(size_t)row * K + k];
    p.h[1] = (_Float16)W[(size_t)row * K + k + 1];
    Wf[idx] = p.u;
}

// ---------------------------------------------------------------------------
// GEMM1 + sigmoid + (a,b) + per-chunk affine summary (scan pass 1, fused).
// Block: 128 threads (4 waves). Tile: 256 rows x 32 frames (= 1 scan chunk).
// B tile staged in LDS in *fragment order*: [ks][half][col][j] uints.
// ---------------------------------------------------------------------------
__global__ __launch_bounds__(128)
void k_gemm1_act(const float* __restrict__ inputs,
                 const unsigned* __restrict__ W1f,
                 const float* __restrict__ B1,
                 float* __restrict__ Aarr, float* __restrict__ Barr,
                 float* __restrict__ Ca,   float* __restrict__ Cb) {
    __shared__ __align__(16) unsigned ldsF[16 * 2 * FPB * 8];  // 32KB
    __shared__ float hs[H_ROWS * 33];                          // padded
    const int tid    = threadIdx.x;
    const int lane   = tid & 31;
    const int wv     = tid >> 5;
    const int half   = lane >> 4;
    const int ln16   = lane & 15;
    const int frame0 = blockIdx.x * FPB;

    // Stage input tile (512 x 32) as f16 pairs directly in fragment order.
    for (int i = tid; i < (INPUT_LEN / 2) * FPB; i += 128) {
        int n  = i & 31;
        int k  = (i >> 5) * 2;
        int ks = k >> 5, kk = k & 31;
        int h, j; kinv(kk, h, j);
        union { _Float16 hh[2]; unsigned u; } p;
        p.hh[0] = (_Float16)inputs[(size_t)k * N_FRAMES + frame0 + n];
        p.hh[1] = (_Float16)inputs[(size_t)(k + 1) * N_FRAMES + frame0 + n];
        ldsF[(((ks * 2 + h) * FPB) + n) * 8 + j] = p.u;
    }
    __syncthreads();

    v8f acc[4][2] = {};
    #pragma unroll 2
    for (int ks = 0; ks < 16; ++ks) {
        FragU bf[2];
        #pragma unroll
        for (int ct = 0; ct < 2; ++ct) {
            int base = ((ks * 2 + half) * FPB + ct * 16 + ln16) * 8;
            bf[ct].q[0] = *(const uint4*)&ldsF[base];
            bf[ct].q[1] = *(const uint4*)&ldsF[base + 4];
        }
        #pragma unroll
        for (int tt = 0; tt < 4; ++tt) {
            int rt = wv * 4 + tt;
            const uint4* ap = (const uint4*)(W1f + ((size_t)(rt * 16 + ks) * 32 + lane) * 8);
            FragU af;
            af.q[0] = ap[0];
            af.q[1] = ap[1];
            acc[tt][0] = __builtin_amdgcn_wmma_f32_16x16x32_f16(
                false, af.v, false, bf[0].v, (short)0, acc[tt][0], false, false);
            acc[tt][1] = __builtin_amdgcn_wmma_f32_16x16x32_f16(
                false, af.v, false, bf[1].v, (short)0, acc[tt][1], false, false);
        }
    }

    // bias + sigmoid -> LDS (256 rows x 32 frames)
    #pragma unroll
    for (int tt = 0; tt < 4; ++tt) {
        int rowbase = wv * 64 + tt * 16;
        #pragma unroll
        for (int ct = 0; ct < 2; ++ct) {
            #pragma unroll
            for (int r = 0; r < 8; ++r) {
                int row = rowbase + r + 8 * half;
                hs[row * 33 + ct * 16 + ln16] = sigmoidf_(acc[tt][ct][r] + B1[row]);
            }
        }
    }
    __syncthreads();

    // a = left*right, b = 1-left; frame-major [frame][m] (512B coalesced)
    // + compose this chunk's affine map (aa, bb) in frame order.
    int m = tid;                 // 0..127 == channel
    float aa = 1.0f, bb = 0.0f;
    for (int n = 0; n < FPB; ++n) {
        float l = hs[m * 33 + n];
        float r = hs[(m + N_MEMORY) * 33 + n];
        float a = l * r;
        float b = 1.0f - l;
        size_t o = (size_t)(frame0 + n) * N_MEMORY + m;
        Aarr[o] = a;
        Barr[o] = b;
        aa = a * aa;
        bb = a * bb + b;
    }
    Ca[blockIdx.x * N_MEMORY + m] = aa;
    Cb[blockIdx.x * N_MEMORY + m] = bb;
}

// ---------------------------------------------------------------------------
// Carry scan, level 1: per group of 32 chunks -> exclusive local prefixes
// (LCa, LCb) and group totals (GSa, GSb).
// ---------------------------------------------------------------------------
__global__ __launch_bounds__(128)
void k_carry_group(const float* __restrict__ Ca, const float* __restrict__ Cb,
                   float* __restrict__ LCa, float* __restrict__ LCb,
                   float* __restrict__ GSa, float* __restrict__ GSb) {
    int g = blockIdx.x, m = threadIdx.x;
    float aa = 1.0f, bb = 0.0f;
    #pragma unroll 8
    for (int i = 0; i < GRP_CH; ++i) {
        size_t c = (size_t)(g * GRP_CH + i) * N_MEMORY + m;
        LCa[c] = aa;
        LCb[c] = bb;
        float a = Ca[c];
        float b = Cb[c];
        aa = a * aa;
        bb = a * bb + b;
    }
    GSa[g * N_MEMORY + m] = aa;
    GSb[g * N_MEMORY + m] = bb;
}

// ---------------------------------------------------------------------------
// Carry scan, level 2: scan the 64 group totals -> state at each group start.
// ---------------------------------------------------------------------------
__global__ __launch_bounds__(128)
void k_carry_top(const float* __restrict__ GSa, const float* __restrict__ GSb,
                 float* __restrict__ Sg) {
    int m = threadIdx.x;
    float s = 0.0f;
    #pragma unroll 8
    for (int g = 0; g < N_GROUPS; ++g) {
        Sg[g * N_MEMORY + m] = s;
        s = GSa[g * N_MEMORY + m] * s + GSb[g * N_MEMORY + m];
    }
}

// ---------------------------------------------------------------------------
// GEMM2 fused with scan pass 3: regenerate z for this block's 32 frames from
// the chunk carry + (a,b), stage as f16 fragments in LDS, WMMA, 1-sigmoid.
// z-tile LDS fragment layout: halves at ((((ks*2+h)*FPB + n)*8 + j)*2 + pos).
// ---------------------------------------------------------------------------
__global__ __launch_bounds__(128)
void k_gemm2_out(const float* __restrict__ Aarr, const float* __restrict__ Barr,
                 const float* __restrict__ LCa,  const float* __restrict__ LCb,
                 const float* __restrict__ Sg,
                 const unsigned* __restrict__ W2f,
                 const float* __restrict__ B2,
                 float* __restrict__ out) {
    __shared__ __align__(16) _Float16 ldsZf[4 * 2 * FPB * 8 * 2];  // 8KB
    __shared__ float hs[H_ROWS * 33];                              // padded
    const int tid    = threadIdx.x;
    const int lane   = tid & 31;
    const int wv     = tid >> 5;
    const int half   = lane >> 4;
    const int ln16   = lane & 15;
    const int c      = blockIdx.x;           // chunk index
    const int frame0 = c * FPB;
    const int g      = c / GRP_CH;

    // z at chunk start, then local recurrence; stage z as f16 fragments.
    {
        int m   = tid;                       // channel == K index of GEMM2
        int ks  = m >> 5, kk = m & 31;
        int kkb = kk & 30, pos = kk & 1;
        int h, j; kinv(kkb, h, j);
        int base = ((ks * 2 + h) * FPB) * 16 + j * 2 + pos;   // + n*16 per frame
        float z = LCa[(size_t)c * N_MEMORY + m] * Sg[g * N_MEMORY + m]
                + LCb[(size_t)c * N_MEMORY + m];
        for (int n = 0; n < FPB; ++n) {
            ldsZf[base + n * 16] = (_Float16)z;
            size_t o = (size_t)(frame0 + n) * N_MEMORY + m;
            z = Aarr[o] * z + Barr[o];
        }
    }
    __syncthreads();
    const unsigned* zf = (const unsigned*)ldsZf;

    v8f acc[4][2] = {};
    #pragma unroll 2
    for (int ks = 0; ks < 4; ++ks) {
        FragU bf[2];
        #pragma unroll
        for (int ct = 0; ct < 2; ++ct) {
            int base = ((ks * 2 + half) * FPB + ct * 16 + ln16) * 8;
            bf[ct].q[0] = *(const uint4*)&zf[base];
            bf[ct].q[1] = *(const uint4*)&zf[base + 4];
        }
        #pragma unroll
        for (int tt = 0; tt < 4; ++tt) {
            int rt = wv * 4 + tt;
            const uint4* ap = (const uint4*)(W2f + ((size_t)(rt * 4 + ks) * 32 + lane) * 8);
            FragU af;
            af.q[0] = ap[0];
            af.q[1] = ap[1];
            acc[tt][0] = __builtin_amdgcn_wmma_f32_16x16x32_f16(
                false, af.v, false, bf[0].v, (short)0, acc[tt][0], false, false);
            acc[tt][1] = __builtin_amdgcn_wmma_f32_16x16x32_f16(
                false, af.v, false, bf[1].v, (short)0, acc[tt][1], false, false);
        }
    }

    // bias + (1 - sigmoid) -> LDS, then float4-coalesced stores.
    #pragma unroll
    for (int tt = 0; tt < 4; ++tt) {
        int rowbase = wv * 64 + tt * 16;
        #pragma unroll
        for (int ct = 0; ct < 2; ++ct) {
            #pragma unroll
            for (int r = 0; r < 8; ++r) {
                int row = rowbase + r + 8 * half;
                hs[row * 33 + ct * 16 + ln16] =
                    1.0f - sigmoidf_(acc[tt][ct][r] + B2[row]);
            }
        }
    }
    __syncthreads();

    for (int i = tid; i < H_ROWS * (FPB / 4); i += 128) {   // 2048 float4s
        int row = i >> 3, seg = i & 7;
        float4 v = make_float4(hs[row * 33 + seg * 4 + 0],
                               hs[row * 33 + seg * 4 + 1],
                               hs[row * 33 + seg * 4 + 2],
                               hs[row * 33 + seg * 4 + 3]);
        *(float4*)(out + (size_t)row * N_FRAMES + frame0 + seg * 4) = v;
    }
}

// ---------------------------------------------------------------------------
extern "C" void kernel_launch(void* const* d_in, const int* in_sizes, int n_in,
                              void* d_out, int out_size, void* d_ws, size_t ws_size,
                              hipStream_t stream) {
    const float* inputs = (const float*)d_in[0];   // (512, 65536)
    const float* W1     = (const float*)d_in[1];   // (256, 512)
    const float* B1     = (const float*)d_in[2];   // (256, 1)
    const float* W2     = (const float*)d_in[3];   // (256, 128)
    const float* B2     = (const float*)d_in[4];   // (256, 1)
    float* out          = (float*)d_out;           // (256, 65536)

    char* ws = (char*)d_ws;
    const size_t W1F_E = 16 * 16 * 32 * 8;               // 65536 uints
    const size_t W2F_E = 16 * 4 * 32 * 8;                // 16384 uints
    const size_t AB_E  = (size_t)N_FRAMES * N_MEMORY;    // 8388608 floats
    const size_t CH_E  = (size_t)N_CHUNKS * N_MEMORY;    // 262144 floats
    const size_t GR_E  = (size_t)N_GROUPS * N_MEMORY;    // 8192 floats

    unsigned* W1f = (unsigned*)ws;  ws += W1F_E * 4;
    unsigned* W2f = (unsigned*)ws;  ws += W2F_E * 4;
    float* Aarr   = (float*)ws;     ws += AB_E * 4;
    float* Barr   = (float*)ws;     ws += AB_E * 4;
    float* Ca     = (float*)ws;     ws += CH_E * 4;
    float* Cb     = (float*)ws;     ws += CH_E * 4;
    float* LCa    = (float*)ws;     ws += CH_E * 4;
    float* LCb    = (float*)ws;     ws += CH_E * 4;
    float* GSa    = (float*)ws;     ws += GR_E * 4;
    float* GSb    = (float*)ws;     ws += GR_E * 4;
    float* Sgc    = (float*)ws;     ws += GR_E * 4;

    // 1) Weights -> WMMA fragment order (f16)
    k_w_to_frag<<<(int)(W1F_E / 256), 256, 0, stream>>>(W1, W1f, INPUT_LEN, 16, (int)W1F_E);
    k_w_to_frag<<<(int)(W2F_E / 256), 256, 0, stream>>>(W2, W2f, N_MEMORY, 4, (int)W2F_E);

    // 2) GEMM1 + sigmoid + (a,b) + chunk summaries
    k_gemm1_act<<<N_CHUNKS, 128, 0, stream>>>(inputs, W1f, B1, Aarr, Barr, Ca, Cb);

    // 3) Hierarchical carry scan over 2048 chunks
    k_carry_group<<<N_GROUPS, 128, 0, stream>>>(Ca, Cb, LCa, LCb, GSa, GSb);
    k_carry_top<<<1, 128, 0, stream>>>(GSa, GSb, Sgc);

    // 4) GEMM2 (+ fused local z regeneration) + output activation
    k_gemm2_out<<<N_CHUNKS, 128, 0, stream>>>(Aarr, Barr, LCa, LCb, Sgc, W2f, B2, out);
}
